// TtCausalSelfAttention_60284160967318
// MI455X (gfx1250) — matrix operand order repaired and checked by reading, not compiled, not run on hardware
//
#include <hip/hip_runtime.h>
#include <stdint.h>

// ---------------------------------------------------------------------------
// CDNA5 (gfx1250) causal self-attention: bf16 WMMA matmuls, f32 softmax,
// TDM (tensor_load_to_lds) tile movement for attention K/V tiles.
// ---------------------------------------------------------------------------

typedef __attribute__((ext_vector_type(16))) __bf16 bf16x16;
typedef __attribute__((ext_vector_type(8)))  float  f32x8;

#define LOG2E 1.44269504088896340736f

union AFrag { bf16x16 v; uint4 q[2]; uint16_t u[16]; };
union CFrag { f32x8 v;   float f[8]; };

__device__ __forceinline__ uint16_t f32_to_bf16(float f) {
  uint32_t u = __builtin_bit_cast(uint32_t, f);
  uint32_t r = u + 0x7FFFu + ((u >> 16) & 1u);   // round-to-nearest-even
  return (uint16_t)(r >> 16);
}

// ---------------------------------------------------------------------------
// TDM: 2D tile load Global -> LDS via Tensor Data Mover.
// D# packing per CDNA5 ISA ch.8 (group0 128b, group1 256b), data_size = 4B.
// tensor_d0_dw / stride_dw / tile_d0_dw are in DWORD units.
// ---------------------------------------------------------------------------
__device__ __forceinline__ void tdm_load_tile_2d(uint32_t lds_byte_addr,
                                                 const void* gsrc,
                                                 uint32_t tensor_d0_dw,
                                                 uint32_t tensor_d1,
                                                 uint32_t stride_dw,
                                                 uint32_t tile_d0_dw,
                                                 uint32_t tile_d1) {
  typedef uint32_t u32x4 __attribute__((ext_vector_type(4)));
  typedef int      i32x4 __attribute__((ext_vector_type(4)));
  typedef int      i32x8 __attribute__((ext_vector_type(8)));
  const uint64_t ga = (uint64_t)(uintptr_t)gsrc;

  u32x4 g0;
  g0[0] = 1u;                                   // count=1 (valid), user mode
  g0[1] = lds_byte_addr;                        // lds_addr [63:32]
  g0[2] = (uint32_t)ga;                         // global_addr low
  g0[3] = (uint32_t)((ga >> 32) & 0x01FFFFFFu)  // global_addr high (57-bit)
        | (2u << 30);                           // type=2 ("image")

  i32x8 g1;
  g1[0] = (int)(2u << 16);                      // data_size=2 -> 4 bytes; mask=0
  g1[1] = (int)((tensor_d0_dw & 0xFFFFu) << 16);                 // dim0[15:0]
  g1[2] = (int)((tensor_d0_dw >> 16) | ((tensor_d1 & 0xFFFFu) << 16));
  g1[3] = (int)((tensor_d1 >> 16) | ((tile_d0_dw & 0xFFFFu) << 16));
  g1[4] = (int)(tile_d1 & 0xFFFFu);             // tile_dim1; tile_dim2=0
  g1[5] = (int)stride_dw;                       // tensor_dim0_stride[31:0]
  g1[6] = 0;                                    // stride[47:32], d1stride[15:0]
  g1[7] = 0;

  i32x4 z4; z4[0] = 0; z4[1] = 0; z4[2] = 0; z4[3] = 0;
#if defined(__clang_major__) && (__clang_major__ >= 23)
  i32x8 z8; z8[0]=0; z8[1]=0; z8[2]=0; z8[3]=0; z8[4]=0; z8[5]=0; z8[6]=0; z8[7]=0;
  __builtin_amdgcn_tensor_load_to_lds(g0, g1, z4, z4, z8, 0);
#else
  __builtin_amdgcn_tensor_load_to_lds(g0, g1, z4, z4, 0);
#endif
}

// ---------------------------------------------------------------------------
// f32 -> bf16 conversion (grid-stride)
// ---------------------------------------------------------------------------
__global__ void cvt_f32_bf16(const float* __restrict__ in,
                             uint16_t* __restrict__ out, int n) {
  int i = blockIdx.x * blockDim.x + threadIdx.x;
  int stride = gridDim.x * blockDim.x;
  for (; i < n; i += stride) out[i] = f32_to_bf16(in[i]);
}

// ---------------------------------------------------------------------------
// Tiled bf16 WMMA GEMM:  out[M,N] = A[M,K] @ B[K,N] + bias[N]
// Block tile 128x128, 8 waves, wave tile 32x64 (2x4 WMMA tiles), K-step 32.
// ---------------------------------------------------------------------------
template<bool OUT_BF16>
__global__ __launch_bounds__(256)
void gemm_bf16_wmma(const uint16_t* __restrict__ A, const uint16_t* __restrict__ Bw,
                    const float* __restrict__ bias,
                    float* __restrict__ outF, uint16_t* __restrict__ outB,
                    int M, int N, int K) {
  __shared__ uint16_t lds_a[128 * 32];   // [row][k]    8 KB
  __shared__ uint16_t lds_bt[128 * 32];  // [col][k]    8 KB

  const int tid  = threadIdx.x;
  const int wave = tid >> 5;
  const int lane = tid & 31;
  const int half = lane >> 4;
  const int l16  = lane & 15;
  const int wm   = wave & 3;   // row group of 32 (2 M-tiles)
  const int wn   = wave >> 2;  // col group of 64 (4 N-tiles)
  const int bm   = blockIdx.y;
  const int bn   = blockIdx.x;

  CFrag acc[2][4];
  #pragma unroll
  for (int mi = 0; mi < 2; ++mi)
    #pragma unroll
    for (int ni = 0; ni < 4; ++ni)
      #pragma unroll
      for (int e = 0; e < 8; ++e) acc[mi][ni].f[e] = 0.0f;

  const int nsteps = K >> 5;
  for (int kt = 0; kt < nsteps; ++kt) {
    __syncthreads();
    {  // A tile 128x32: each thread moves 16 contiguous bf16 (32 B)
      int row = tid >> 1, seg = tid & 1;
      const uint4* s4 = (const uint4*)(A + (size_t)(bm * 128 + row) * K + kt * 32 + seg * 16);
      uint4* d4 = (uint4*)&lds_a[row * 32 + seg * 16];
      d4[0] = s4[0]; d4[1] = s4[1];
      if (kt + 1 < nsteps)  // global_prefetch_b8 of next A tile
        __builtin_prefetch(A + (size_t)(bm * 128 + row) * K + (kt + 1) * 32 + seg * 16, 0, 1);
    }
    {  // B tile 32x128, stored transposed: lds_bt[n][k]
      int k = tid >> 3, chunk = tid & 7;
      const uint4* s4 = (const uint4*)(Bw + (size_t)(kt * 32 + k) * N + bn * 128 + chunk * 16);
      uint4 tmp[2]; tmp[0] = s4[0]; tmp[1] = s4[1];
      const uint16_t* t = (const uint16_t*)tmp;
      #pragma unroll
      for (int i = 0; i < 16; ++i) lds_bt[(chunk * 16 + i) * 32 + k] = t[i];
      if (kt + 1 < nsteps)  // global_prefetch_b8 of next B tile
        __builtin_prefetch(Bw + (size_t)((kt + 1) * 32 + k) * N + bn * 128 + chunk * 16, 0, 1);
    }
    __syncthreads();

    AFrag a[2], b[4];
    #pragma unroll
    for (int mi = 0; mi < 2; ++mi) {   // A 16x32 fragment (ISA 16-bit A layout)
      const uint16_t* base = &lds_a[(wm * 32 + mi * 16 + l16) * 32];
      a[mi].q[0] = *(const uint4*)(base + half * 8);
      a[mi].q[1] = *(const uint4*)(base + 16 + half * 8);
    }
    #pragma unroll
    for (int ni = 0; ni < 4; ++ni) {   // B 32x16 fragment (ISA 16-bit layout, same k order as A)
      const uint16_t* base = &lds_bt[(wn * 64 + ni * 16 + l16) * 32];
      b[ni].q[0] = *(const uint4*)(base + half * 8);
      b[ni].q[1] = *(const uint4*)(base + 16 + half * 8);
    }
    #pragma unroll
    for (int mi = 0; mi < 2; ++mi)
      #pragma unroll
      for (int ni = 0; ni < 4; ++ni)
        acc[mi][ni].v = __builtin_amdgcn_wmma_f32_16x16x32_bf16(
            false, a[mi].v, false, b[ni].v, (short)0, acc[mi][ni].v, false, false);
  }

  #pragma unroll
  for (int mi = 0; mi < 2; ++mi)
    #pragma unroll
    for (int ni = 0; ni < 4; ++ni) {
      int col = bn * 128 + wn * 64 + ni * 16 + l16;
      float bv = bias[col];
      #pragma unroll
      for (int r = 0; r < 8; ++r) {
        int row = bm * 128 + wm * 32 + mi * 16 + r + half * 8;
        float v = acc[mi][ni].f[r] + bv;
        if (OUT_BF16) outB[(size_t)row * N + col] = f32_to_bf16(v);
        else          outF[(size_t)row * N + col] = v;
      }
    }
}

// ---------------------------------------------------------------------------
// Flash-style causal attention over bf16 qkv ([B*T, 3C], q|k|v, head h at h*64).
// One workgroup per (qb, h, b): 128 q-rows, 8 waves x 16 rows each.
// K/V tiles fetched with the Tensor Data Mover; online softmax in f32.
// ---------------------------------------------------------------------------
__global__ __launch_bounds__(256)
void attn_wmma(const uint16_t* __restrict__ qkv, uint16_t* __restrict__ y) {
  constexpr int T = 1024, C = 1024, HD = 64, QB = 128, KB = 64;
  __shared__ uint16_t k_lds[KB * HD];      // [key][d]        8 KB  (TDM dest)
  __shared__ uint16_t v_lds[KB * HD];      // [key][d]        8 KB  (TDM dest)
  __shared__ uint16_t vT_lds[HD * KB];     // [d][key]        8 KB
  __shared__ uint16_t p_lds[8 * 16 * KB];  // per-wave P     16 KB

  const int tid = threadIdx.x;
  const int wave = tid >> 5, lane = tid & 31;
  const int half = lane >> 4, l16 = lane & 15;
  const int qb = blockIdx.x, h = blockIdx.y, b = blockIdx.z;

  const size_t rs = 3 * C;  // qkv row stride (elements)
  const uint16_t* qbase = qkv + (size_t)(b * T) * rs + h * HD;
  const uint16_t* kbase = qbase + C;
  const uint16_t* vbase = qbase + 2 * C;

  const int qrow0 = qb * QB + wave * 16;

  AFrag aq[2];  // Q 16x64 as two 16x32 A-fragments
  #pragma unroll
  for (int j = 0; j < 2; ++j) {
    const uint16_t* src = qbase + (size_t)(qrow0 + l16) * rs + j * 32;
    aq[j].q[0] = *(const uint4*)(src + half * 8);
    aq[j].q[1] = *(const uint4*)(src + 16 + half * 8);
  }

  CFrag acc[4];
  float run_m[8], run_l[8];
  #pragma unroll
  for (int t = 0; t < 4; ++t)
    #pragma unroll
    for (int e = 0; e < 8; ++e) acc[t].f[e] = 0.0f;
  #pragma unroll
  for (int r = 0; r < 8; ++r) { run_m[r] = -1e30f; run_l[r] = 0.0f; }

  const uint32_t k_lds_addr = (uint32_t)(uintptr_t)&k_lds[0];
  const uint32_t v_lds_addr = (uint32_t)(uintptr_t)&v_lds[0];

  const int nkb = 2 * qb + 2;  // key blocks covering keys <= qb*128+127
  for (int kb = 0; kb < nkb; ++kb) {
    __syncthreads();   // previous iteration finished with k_lds / v_lds
    if (wave == 0) {   // one wave drives the Tensor Data Mover
      const uint16_t* kblk = kbase + (size_t)(kb * KB) * rs;
      const uint16_t* vblk = vbase + (size_t)(kb * KB) * rs;
      // 2D tile: 32 dwords (64 bf16) x 64 rows, row stride 1536 dwords.
      tdm_load_tile_2d(k_lds_addr, kblk, 1536u, 64u, 1536u, 32u, 64u);
      tdm_load_tile_2d(v_lds_addr, vblk, 1536u, 64u, 1536u, 32u, 64u);
      __builtin_amdgcn_s_wait_tensorcnt(0);
    }
    __syncthreads();   // tiles visible to all waves
    {  // transpose V in LDS: vT[d][key] = v[key][d]
      int row = tid >> 2, seg = tid & 3;
      uint4 tmp[2];
      const uint4* vsrc = (const uint4*)&v_lds[row * HD + seg * 16];
      tmp[0] = vsrc[0]; tmp[1] = vsrc[1];
      const uint16_t* tt = (const uint16_t*)tmp;
      #pragma unroll
      for (int i = 0; i < 16; ++i) vT_lds[(seg * 16 + i) * KB + row] = tt[i];
    }
    __syncthreads();

    if (kb * KB <= qrow0 + 15) {  // wave-uniform: skip fully-masked blocks
      // ---- S = Q K^T (4 key sub-tiles of 16) ----
      CFrag s[4];
      #pragma unroll
      for (int nt = 0; nt < 4; ++nt) {
        #pragma unroll
        for (int e = 0; e < 8; ++e) s[nt].f[e] = 0.0f;
        #pragma unroll
        for (int j = 0; j < 2; ++j) {
          AFrag bk;  // B(kk,n) = K[key n][hd kk]
          const uint16_t* base = &k_lds[(nt * 16 + l16) * HD + j * 32];
          bk.q[0] = *(const uint4*)(base + half * 8);
          bk.q[1] = *(const uint4*)(base + 16 + half * 8);
          s[nt].v = __builtin_amdgcn_wmma_f32_16x16x32_bf16(
              false, aq[j].v, false, bk.v, (short)0, s[nt].v, false, false);
        }
      }
      // ---- scale + causal mask + block row-max ----
      float bmax[8];
      #pragma unroll
      for (int r = 0; r < 8; ++r) bmax[r] = -1e30f;
      #pragma unroll
      for (int nt = 0; nt < 4; ++nt) {
        int key = kb * KB + nt * 16 + l16;
        #pragma unroll
        for (int r = 0; r < 8; ++r) {
          int qr = qrow0 + r + half * 8;
          float v = s[nt].f[r] * 0.125f;          // 1/sqrt(64)
          v = (key <= qr) ? v : -1e30f;
          s[nt].f[r] = v;
          bmax[r] = fmaxf(bmax[r], v);
        }
      }
      #pragma unroll
      for (int r = 0; r < 8; ++r) {               // reduce within 16-lane half
        float v = bmax[r];
        #pragma unroll
        for (int m = 1; m < 16; m <<= 1) v = fmaxf(v, __shfl_xor(v, m, 32));
        bmax[r] = v;
      }
      float alpha[8];
      #pragma unroll
      for (int r = 0; r < 8; ++r) {
        float nm = fmaxf(run_m[r], bmax[r]);
        alpha[r] = exp2f((run_m[r] - nm) * LOG2E);
        run_m[r] = nm;
      }
      // ---- P = exp(S - m), row sums ----
      float bsum[8];
      #pragma unroll
      for (int r = 0; r < 8; ++r) bsum[r] = 0.0f;
      #pragma unroll
      for (int nt = 0; nt < 4; ++nt)
        #pragma unroll
        for (int r = 0; r < 8; ++r) {
          float p = exp2f((s[nt].f[r] - run_m[r]) * LOG2E);
          s[nt].f[r] = p;
          bsum[r] += p;
        }
      #pragma unroll
      for (int r = 0; r < 8; ++r) {
        float v = bsum[r];
        #pragma unroll
        for (int m = 1; m < 16; m <<= 1) v += __shfl_xor(v, m, 32);
        run_l[r] = run_l[r] * alpha[r] + v;
      }
      #pragma unroll
      for (int t = 0; t < 4; ++t)
        #pragma unroll
        for (int r = 0; r < 8; ++r) acc[t].f[r] *= alpha[r];

      // ---- stage P (bf16) in wave-private LDS, reload as A-fragments ----
      uint16_t* pw = &p_lds[wave * 16 * KB];
      #pragma unroll
      for (int nt = 0; nt < 4; ++nt)
        #pragma unroll
        for (int r = 0; r < 8; ++r)
          pw[(r + half * 8) * KB + nt * 16 + l16] = f32_to_bf16(s[nt].f[r]);

      #pragma unroll
      for (int j = 0; j < 2; ++j) {   // O += P V  (K-dim = keys, 2 steps of 32)
        AFrag ap;
        const uint16_t* base = &pw[l16 * KB + j * 32];
        ap.q[0] = *(const uint4*)(base + half * 8);
        ap.q[1] = *(const uint4*)(base + 16 + half * 8);
        #pragma unroll
        for (int vt = 0; vt < 4; ++vt) {
          AFrag bv;  // B(kk,n) = V[key kk][hd n] = vT[n][kk]
          const uint16_t* vb = &vT_lds[(vt * 16 + l16) * KB + j * 32];
          bv.q[0] = *(const uint4*)(vb + half * 8);
          bv.q[1] = *(const uint4*)(vb + 16 + half * 8);
          acc[vt].v = __builtin_amdgcn_wmma_f32_16x16x32_bf16(
              false, ap.v, false, bv.v, (short)0, acc[vt].v, false, false);
        }
      }
    }
  }

  // ---- normalize and scatter y (bf16, [B,T,C] with heads re-merged) ----
  #pragma unroll
  for (int vt = 0; vt < 4; ++vt)
    #pragma unroll
    for (int r = 0; r < 8; ++r) {
      int qr = qrow0 + r + half * 8;
      float invl = 1.0f / run_l[r];
      y[(size_t)(b * T + qr) * C + h * HD + vt * 16 + l16] =
          f32_to_bf16(acc[vt].f[r] * invl);
    }
}

// ---------------------------------------------------------------------------
// Launch
// ---------------------------------------------------------------------------
extern "C" void kernel_launch(void* const* d_in, const int* in_sizes, int n_in,
                              void* d_out, int out_size, void* d_ws, size_t ws_size,
                              hipStream_t stream) {
  (void)in_sizes; (void)n_in; (void)out_size; (void)ws_size;
  const float* x      = (const float*)d_in[0];
  const float* w_attn = (const float*)d_in[1];
  const float* b_attn = (const float*)d_in[2];
  const float* w_proj = (const float*)d_in[3];
  const float* b_proj = (const float*)d_in[4];

  const int B = 8, T = 1024, C = 1024;
  const int M = B * T;       // 8192
  const int N1 = 3 * C;      // 3072

  char* ws = (char*)d_ws;
  size_t off = 0;
  auto alloc = [&](size_t bytes) {
    void* p = ws + off;
    off += (bytes + 255) & ~(size_t)255;
    return p;
  };
  uint16_t* x_bf   = (uint16_t*)alloc((size_t)M * C * 2);    // 16 MB
  uint16_t* wa_bf  = (uint16_t*)alloc((size_t)C * N1 * 2);   //  6 MB
  uint16_t* wp_bf  = (uint16_t*)alloc((size_t)C * C * 2);    //  2 MB
  uint16_t* qkv_bf = (uint16_t*)alloc((size_t)M * N1 * 2);   // 48 MB
  uint16_t* y_bf   = (uint16_t*)alloc((size_t)M * C * 2);    // 16 MB

  cvt_f32_bf16<<<2048, 256, 0, stream>>>(x, x_bf, M * C);
  cvt_f32_bf16<<<2048, 256, 0, stream>>>(w_attn, wa_bf, C * N1);
  cvt_f32_bf16<<<1024, 256, 0, stream>>>(w_proj, wp_bf, C * C);

  dim3 g1(N1 / 128, M / 128);   // qkv = x @ w_attn + b_attn
  gemm_bf16_wmma<true><<<g1, 256, 0, stream>>>(x_bf, wa_bf, b_attn,
                                               nullptr, qkv_bf, M, N1, C);

  dim3 g2(T / 128, 16, B);      // attention
  attn_wmma<<<g2, 256, 0, stream>>>(qkv_bf, y_bf);

  dim3 g3(C / 128, M / 128);    // out = y @ w_proj + b_proj (f32)
  gemm_bf16_wmma<false><<<g3, 256, 0, stream>>>(y_bf, wp_bf, b_proj,
                                                (float*)d_out, nullptr, M, C, C);
}
